// LSTMLayer_2345052143929
// MI455X (gfx1250) — compile-verified, matrix operand order checked by reading, not run on hardware
//
#include <hip/hip_runtime.h>
#include <math.h>

#define B_ 128
#define T_ 64
#define D_ 512
#define U_ 512
#define O_ 512

typedef __attribute__((ext_vector_type(16))) __bf16 v16bf;
typedef __attribute__((ext_vector_type(8)))  float  v8f;

// LDS B-tile column stride in bf16 elements, padded (34 halves = 17 banks)
#define LDSB_STRIDE 34

// Read one 16x32 bf16 B fragment (ISA 16-bit B layout) from LDS.
__device__ __forceinline__ v16bf load_bfrag(const __bf16* bp, int half)
{
    v16bf b;
    #pragma unroll
    for (int v = 0; v < 8; ++v) {
        int kk = ((v & 3) << 1) + (half << 3) + ((v >> 2) << 4);
        b[2 * v]     = bp[kk];
        b[2 * v + 1] = bp[kk + 1];
    }
    return b;
}

// ---------------------------------------------------------------------------
// Streaming GEMM tile (used by the fully parallel Zx kernel, where 1024 WGs
// hide the per-chunk staging latency by occupancy).
// C[128 x NCOLS] += A[128 x K] * B[K x NCOLS]; col c -> colbase+(c/16)*gstride+(c&15)
// ---------------------------------------------------------------------------
template<int NCOLS>
__device__ __forceinline__ void gemm_tile(
    const float* __restrict__ A0, int lda,
    const float* __restrict__ Bmat, int ldb,
    int colbase, int gstride, int K,
    __bf16* lds_b, v8f* acc)
{
    const int tid  = threadIdx.x;
    const int wid  = tid >> 5;
    const int lane = tid & 31;
    const int half = lane >> 4;   // lanes 16-31 hold K+8 per ISA 16-bit layout
    const int l15  = lane & 15;
    const int mrow = (wid << 4) + l15;
    constexpr int NS = NCOLS / 16;

    for (int k0 = 0; k0 < K; k0 += 32) {
        __syncthreads();
        for (int idx = tid; idx < 32 * NCOLS; idx += 256) {
            int kk = idx / NCOLS;
            int cc = idx - kk * NCOLS;
            int n  = colbase + (cc >> 4) * gstride + (cc & 15);
            float v = __builtin_nontemporal_load(&Bmat[(long)(k0 + kk) * ldb + n]);
            lds_b[cc * LDSB_STRIDE + kk] = (__bf16)v;
        }
        v16bf afrag;
        const float* arow = A0 + (long)mrow * lda + k0;
        #pragma unroll
        for (int v = 0; v < 8; ++v) {
            int kk = ((v & 3) << 1) + (half << 3) + ((v >> 2) << 4);
            float2 f2 = *reinterpret_cast<const float2*>(arow + kk);
            afrag[2 * v]     = (__bf16)f2.x;
            afrag[2 * v + 1] = (__bf16)f2.y;
        }
        __syncthreads();
        // Software-pipelined subtile sweep: fragment s+1 loads overlap WMMA s.
        v16bf bcur = load_bfrag(lds_b + l15 * LDSB_STRIDE, half);
        #pragma unroll
        for (int s = 0; s < NS; ++s) {
            v16bf bnext = bcur;
            if (s + 1 < NS)
                bnext = load_bfrag(lds_b + ((s + 1) * 16 + l15) * LDSB_STRIDE, half);
            acc[s] = __builtin_amdgcn_wmma_f32_16x16x32_bf16(
                false, afrag, false, bcur, (short)0, acc[s], false, false);
            bcur = bnext;
        }
    }
}

// ---------------------------------------------------------------------------
// Latency-optimized resident GEMM for the sequential step kernel (K fixed 512).
//  * whole K x NCOLS weight tile staged to LDS in one deep load burst
//  * all 16 A k-chunks preloaded to VGPRs before the single barrier
//  * optional L2 prefetch of next timestep's weight tile (overlaps compute)
//  * B fragments double-buffered so DS latency hides under WMMA execution
// LDS layout: [chunk][col][kk], col stride 34 halves (17 banks).
// ---------------------------------------------------------------------------
template<int NCOLS>
__device__ __forceinline__ void gemm_resident(
    const float* __restrict__ A0, int lda,
    const float* __restrict__ Bmat, int ldb,
    const float* __restrict__ Bnext,      // next step's tile base (or nullptr)
    int colbase, int gstride,
    __bf16* lds_b, v8f* acc)
{
    const int tid  = threadIdx.x;
    const int wid  = tid >> 5;
    const int lane = tid & 31;
    const int half = lane >> 4;
    const int l15  = lane & 15;
    const int mrow = (wid << 4) + l15;
    constexpr int NS = NCOLS / 16;
    constexpr int NFRAG = 16 * NS;
    const int CHSTRIDE = NCOLS * LDSB_STRIDE;     // halves per k-chunk

    // ---- burst-stage whole 512 x NCOLS fp32 tile -> bf16 LDS (8-deep MLP) ----
    const int c     = tid & (NCOLS - 1);
    const int kbase = tid / NCOLS;
    const int kstep = 256 / NCOLS;
    const int n     = colbase + (c >> 4) * gstride + (c & 15);
    const int nb    = 512 / kstep / 8;
    for (int b2 = 0; b2 < nb; ++b2) {
        float tmp[8];
        #pragma unroll
        for (int j = 0; j < 8; ++j) {
            int k = kbase + (b2 * 8 + j) * kstep;
            tmp[j] = __builtin_nontemporal_load(&Bmat[(long)k * ldb + n]);
        }
        #pragma unroll
        for (int j = 0; j < 8; ++j) {
            int k = kbase + (b2 * 8 + j) * kstep;
            lds_b[(k >> 5) * CHSTRIDE + c * LDSB_STRIDE + (k & 31)] = (__bf16)tmp[j];
        }
    }

    // ---- prefetch next step's weight slice into L2 (1 probe / 64B segment) ---
    if (Bnext) {
        constexpr int G = NCOLS / 16;
        for (int i = tid; i < 512 * G; i += 256) {
            int k = i / G, q = i - (i / G) * G;
            __builtin_prefetch(&Bnext[(long)k * ldb + colbase + q * gstride], 0, 1);
        }
    }

    // ---- preload all 16 A k-chunks for this wave's 16-row strip (128 VGPR) ---
    v16bf afrag[16];
    #pragma unroll
    for (int ch = 0; ch < 16; ++ch) {
        const float* arow = A0 + (long)mrow * lda + ch * 32;
        #pragma unroll
        for (int v = 0; v < 8; ++v) {
            int kk = ((v & 3) << 1) + (half << 3) + ((v >> 2) << 4);
            float2 f2 = *reinterpret_cast<const float2*>(arow + kk);
            afrag[ch][2 * v]     = (__bf16)f2.x;
            afrag[ch][2 * v + 1] = (__bf16)f2.y;
        }
    }
    __syncthreads();

    // ---- pipelined WMMA sweep: fragment i+1 DS loads overlap WMMA i ---------
    v16bf bcur = load_bfrag(lds_b + l15 * LDSB_STRIDE, half);
    #pragma unroll
    for (int i = 0; i < NFRAG; ++i) {
        v16bf bnext = bcur;
        if (i + 1 < NFRAG) {
            int ch1 = (i + 1) / NS, s1 = (i + 1) - ((i + 1) / NS) * NS;
            bnext = load_bfrag(lds_b + ch1 * CHSTRIDE
                               + (s1 * 16 + l15) * LDSB_STRIDE, half);
        }
        int ch = i / NS, s = i - (i / NS) * NS;
        acc[s] = __builtin_amdgcn_wmma_f32_16x16x32_bf16(
            false, afrag[ch], false, bcur, (short)0, acc[s], false, false);
        bcur = bnext;
    }
}

// Kernel 1: Zx[t][m][n] = x[m][t][:] @ Wx[t][:, n] + b[t][n], all T in parallel.
__global__ __launch_bounds__(256) void zx_kernel(
    const float* __restrict__ x, const float* __restrict__ Wx,
    const float* __restrict__ bias, float* __restrict__ zx)
{
    __shared__ __bf16 lds_b[128 * LDSB_STRIDE];
    const int t  = blockIdx.y;
    const int n0 = blockIdx.x * 128;
    v8f acc[8] = {};
    gemm_tile<128>(x + (long)t * D_, T_ * D_,
                   Wx + (long)t * D_ * (4 * U_), 4 * U_,
                   n0, 16, D_, lds_b, acc);

    const int tid = threadIdx.x, wid = tid >> 5, lane = tid & 31;
    const int half = lane >> 4, l15 = lane & 15;
    #pragma unroll
    for (int s = 0; s < 8; ++s) {
        int n = n0 + s * 16 + l15;
        float bv = bias[t * (4 * U_) + n];
        #pragma unroll
        for (int r = 0; r < 8; ++r) {
            int m = (wid << 4) + (half << 3) + r;
            zx[((long)(t * B_ + m)) * (4 * U_) + n] = acc[s][r] + bv;
        }
    }
}

// Kernel 2 (launched T+1 times, stream-ordered = recurrence sync):
//  blocks 0..31 : z_t = Zx[t] + h_{t-1} @ Wh_t (gate-grouped; per-lane cell
//                 update), writes c_t in place + h_t double-buffered.
//  blocks 32..63: y_{t-1} = h_{t-1} @ Wy_{t-1} + by_{t-1} -> out.
__global__ __launch_bounds__(256) void step_kernel(
    int t,
    const float* __restrict__ Wh, const float* __restrict__ Wy,
    const float* __restrict__ by, const float* __restrict__ zx,
    float* __restrict__ cbuf, const float* __restrict__ hprev,
    float* __restrict__ hnext, float* __restrict__ out)
{
    __shared__ __bf16 lds_b[16 * 64 * LDSB_STRIDE];   // 69632 B of 320 KB WGP LDS
    const int tid = threadIdx.x, wid = tid >> 5, lane = tid & 31;
    const int half = lane >> 4, l15 = lane & 15;

    if (blockIdx.x < 32) {
        if (t >= T_) return;                       // last launch: y only
        const int u0 = blockIdx.x * 16;            // 16 u-columns per WG
        const float* Wh_t = Wh + (long)t * U_ * (4 * U_);
        const float* Wh_n = (t + 1 < T_) ? Wh_t + (long)U_ * (4 * U_) : nullptr;
        v8f acc[4] = {};                           // s = gate q, cols u0 + q*512
        gemm_resident<64>(hprev, U_, Wh_t, 4 * U_, Wh_n, u0, U_, lds_b, acc);
        const int u = u0 + l15;
        #pragma unroll
        for (int r = 0; r < 8; ++r) {
            int m = (wid << 4) + (half << 3) + r;
            long zoff = ((long)(t * B_ + m)) * (4 * U_);
            float zi = acc[0][r] + zx[zoff + u];
            float zf = acc[1][r] + zx[zoff + U_ + u];
            float zg = acc[2][r] + zx[zoff + 2 * U_ + u];
            float zo = acc[3][r] + zx[zoff + 3 * U_ + u];
            float ig = 1.f / (1.f + __expf(-zi));
            float fg = 1.f / (1.f + __expf(-zf));
            float gg = tanhf(zg);
            float og = 1.f / (1.f + __expf(-zo));
            float cc = fg * cbuf[m * U_ + u] + ig * gg;
            cbuf[m * U_ + u]  = cc;
            hnext[m * U_ + u] = og * tanhf(cc);
        }
    } else {
        if (t == 0) return;                        // no h yet
        const int ty = t - 1;
        const int n0 = (blockIdx.x - 32) * 16;
        const float* Wy_t = Wy + (long)ty * U_ * O_;
        const float* Wy_n = (t < T_) ? Wy_t + (long)U_ * O_ : nullptr;
        v8f acc[1] = {};
        gemm_resident<16>(hprev, U_, Wy_t, O_, Wy_n, n0, 16, lds_b, acc);
        const int n = n0 + l15;
        float bv = by[ty * O_ + n];
        #pragma unroll
        for (int r = 0; r < 8; ++r) {
            int m = (wid << 4) + (half << 3) + r;
            out[((long)m * T_ + ty) * O_ + n] = acc[0][r] + bv;
        }
    }
}

extern "C" void kernel_launch(void* const* d_in, const int* in_sizes, int n_in,
                              void* d_out, int out_size, void* d_ws, size_t ws_size,
                              hipStream_t stream) {
    (void)in_sizes; (void)n_in; (void)out_size; (void)ws_size;
    const float* x  = (const float*)d_in[0];
    const float* c0 = (const float*)d_in[1];
    const float* h0 = (const float*)d_in[2];
    const float* Wx = (const float*)d_in[3];
    const float* Wh = (const float*)d_in[4];
    const float* bb = (const float*)d_in[5];
    const float* Wy = (const float*)d_in[6];
    const float* by = (const float*)d_in[7];
    float* out = (float*)d_out;

    // workspace: Zx (64 MB, L2-resident on a 192 MB L2) + c + double-buffered h
    float* zx    = (float*)d_ws;
    float* cbuf  = zx + (size_t)T_ * B_ * 4 * U_;
    float* hbuf0 = cbuf + (size_t)B_ * U_;
    float* hbuf1 = hbuf0 + (size_t)B_ * U_;

    hipMemcpyAsync(cbuf,  c0, (size_t)B_ * U_ * sizeof(float),
                   hipMemcpyDeviceToDevice, stream);
    hipMemcpyAsync(hbuf0, h0, (size_t)B_ * U_ * sizeof(float),
                   hipMemcpyDeviceToDevice, stream);

    zx_kernel<<<dim3(16, T_), 256, 0, stream>>>(x, Wx, bb, zx);

    for (int t = 0; t <= T_; ++t) {
        const float* hp = (t & 1) ? hbuf1 : hbuf0;
        float*       hn = (t & 1) ? hbuf0 : hbuf1;
        step_kernel<<<64, 256, 0, stream>>>(t, Wh, Wy, by, zx, cbuf, hp, hn, out);
    }
}
